// GRUODEDecay_50517405335821
// MI455X (gfx1250) — matrix operand order repaired and checked by reading, not compiled, not run on hardware
//
#include <hip/hip_runtime.h>
#include <hip/hip_bf16.h>
#include <math.h>

// GRU-ODE decay: B=64, T=32, I=H=256.
// Single persistent workgroup: 512 threads = 16 wave32 waves on one WGP.
// All GEMMs via v_wmma_f32_16x16x32_bf16. f32 ODE state in LDS, W1 cached in LDS,
// W_ih/W_hh/W2/W3 streamed as bf16 from L2 with depth-2 B-tile pipelining.

#define B_  64
#define T_  32
#define H_  256
#define NT  512  // threads per block
#define NW  16   // waves per block

typedef __attribute__((ext_vector_type(16))) __bf16 v16bf;
typedef __attribute__((ext_vector_type(8)))  __bf16 v8bf;
typedef __attribute__((ext_vector_type(8)))  float  v8f;

// fast device math: native v_exp/v_log/v_rcp (VALU co-executes with XDL WMMA;
// error well below the bf16 GEMM quantization floor)
__device__ __forceinline__ float sigmoidf_(float x) {
    return __builtin_amdgcn_rcpf(1.0f + __expf(-x));
}
__device__ __forceinline__ float tanhf_(float x) {
    // tanh(x) = 1 - 2/(exp(2x)+1); exact at both saturation ends
    return 1.0f - 2.0f * __builtin_amdgcn_rcpf(__expf(2.0f * x) + 1.0f);
}
// jax.nn.softplus = log1p(exp(x)); guarded native form
__device__ __forceinline__ float softplusf_(float x) {
    return (x > 20.0f) ? x : __logf(1.0f + __expf(x));
}

__device__ __forceinline__ v8f wmma_bf16(v16bf a, v16bf b, v8f c) {
    return __builtin_amdgcn_wmma_f32_16x16x32_bf16(
        /*neg_a=*/false, a, /*neg_b=*/false, b,
        /*c_mod=*/(short)0, c, /*reuse_a=*/false, /*reuse_b=*/false);
}

// A operand (16x32 bf16 tile) from a row-major [*][256] bf16 buffer (LDS).
// Contiguous k order (same as load_b): lanes 0-15 hold M=lane, K = kb*32 + 0..15;
//             lanes 16-31 hold M=lane-16, K = kb*32 + 16..31.
__device__ __forceinline__ v16bf load_a(const __bf16* abuf, int mrow, int kb, int khalf) {
    const __bf16* p = abuf + mrow * H_ + kb * 32 + khalf * 16;
    v8bf c0 = *(const v8bf*)(p);
    v8bf c1 = *(const v8bf*)(p + 8);
    v16bf a;
#pragma unroll
    for (int i = 0; i < 8; ++i) { a[i] = c0[i]; a[i + 8] = c1[i]; }
    return a;
}

// Same A layout but sourced from a row-major f32 buffer with on-the-fly bf16 convert.
__device__ __forceinline__ v16bf load_a_f32(const float* ybuf, int mrow, int kb, int khalf) {
    const float* p = ybuf + mrow * H_ + kb * 32 + khalf * 16;
    v16bf a;
#pragma unroll
    for (int i = 0; i < 8; ++i) { a[i] = (__bf16)p[i]; a[i + 8] = (__bf16)p[i + 8]; }
    return a;
}

// B operand (32x16 bf16 tile) from row-major weights W[nrows][256] (row = output col).
// Contiguous k order (same as load_a): lanes 0-15 hold N=lane, K = kb*32 + 0..15;
//             lanes 16-31 hold N=lane-16, K = kb*32 + 16..31.  (32B contiguous load)
__device__ __forceinline__ v16bf load_b(const __bf16* W, int nrow, int kb, int khalf) {
    return *(const v16bf*)(W + (size_t)nrow * H_ + kb * 32 + khalf * 16);
}

// One 64x16 output strip of C = A(64x256) @ W^T, K fully reduced.
// Depth-2 software pipeline on the B tile (one 32B load in flight across 2 kb steps).
template <bool AF32>
__device__ __forceinline__ void gemm_strip(const void* Abuf, const __bf16* W,
                                           int lcol, int khalf, int col, v8f acc[4]) {
    v16bf b0 = load_b(W, col, 0, khalf);
    v16bf b1 = load_b(W, col, 1, khalf);
#pragma clang loop unroll_count(2)
    for (int kb = 0; kb < 8; ++kb) {
        v16bf bn = load_b(W, col, (kb + 2) & 7, khalf);
#pragma unroll
        for (int m = 0; m < 4; ++m) {
            v16bf a;
            if constexpr (AF32) a = load_a_f32((const float*)Abuf, m * 16 + lcol, kb, khalf);
            else                a = load_a((const __bf16*)Abuf, m * 16 + lcol, kb, khalf);
            acc[m] = wmma_bf16(a, b0, acc[m]);
        }
        b0 = b1;
        b1 = bn;
    }
}

__global__ void convert_weights_kernel(const float* __restrict__ Wih,
                                       const float* __restrict__ Whh,
                                       const float* __restrict__ W1,
                                       const float* __restrict__ W2,
                                       const float* __restrict__ W3,
                                       __bf16* __restrict__ dst) {
    int i = blockIdx.x * blockDim.x + threadIdx.x;
    const int n_ih = 768 * 256;      // 196608
    const int n_sq = 256 * 256;      // 65536
    float v;
    if (i < n_ih)                       v = Wih[i];
    else if (i < 2 * n_ih)              v = Whh[i - n_ih];
    else if (i < 2 * n_ih + n_sq)       v = W1[i - 2 * n_ih];
    else if (i < 2 * n_ih + 2 * n_sq)   v = W2[i - 2 * n_ih - n_sq];
    else if (i < 2 * n_ih + 3 * n_sq)   v = W3[i - 2 * n_ih - 2 * n_sq];
    else return;
    dst[i] = (__bf16)v;
}

__global__ __launch_bounds__(NT, 1)
void gru_ode_kernel(const float* __restrict__ input,   // (B,T,I) f32
                    const float* __restrict__ times,   // (B,T)   f32
                    const float* __restrict__ b_ih, const float* __restrict__ b_hh,
                    const float* __restrict__ b1, const float* __restrict__ b2,
                    const float* __restrict__ b3,
                    const __bf16* __restrict__ Wih, const __bf16* __restrict__ Whh,
                    const __bf16* __restrict__ W1,  const __bf16* __restrict__ W2,
                    const __bf16* __restrict__ W3,
                    float* __restrict__ hglob,          // (B,H) f32 scratch
                    float* __restrict__ out)            // (B,T,H) f32
{
    __shared__ float  y[B_][H_];        // 64 KB  f32 ODE state
    __shared__ __bf16 a0[B_][H_];       // 32 KB  bf16 A staging (x_t / stage2 out)
    __shared__ __bf16 a1[B_][H_];       // 32 KB  bf16 A staging (h   / stage1 out)
    __shared__ __bf16 W1lds[H_][H_];    // 128 KB W1 resident in LDS
    __shared__ float  tsrt[B_];
    __shared__ float  dts[B_];
    __shared__ int    order[B_];        // order[rank] = batch row (stable sort)

    const int tid   = threadIdx.x;
    const int lane  = tid & 31;
    const int wave  = tid >> 5;         // 0..15 == N-strip id
    const int khalf = lane >> 4;        // 0: lanes 0-15, 1: lanes 16-31
    const int lcol  = lane & 15;
    const int col   = wave * 16 + lcol; // output column owned by this lane (strips)

    // h0 = 0; stage W1 into LDS
    for (int i = tid; i < B_ * H_; i += NT) hglob[i] = 0.0f;
    for (int i = tid; i < H_ * H_; i += NT) (&W1lds[0][0])[i] = W1[i];
    __syncthreads();

    for (int tstep = 0; tstep < T_; ++tstep) {
        // ---- phase A: time ranks + stage x_t -> a0, h -> a1 ----
        if (tid < B_) {
            float tb = times[tid * T_ + tstep];
            int rk = 0;  // stable-sort rank (tie-break by index)
            for (int j = 0; j < B_; ++j) {
                float tj = times[j * T_ + tstep];
                rk += ((tj < tb) || (tj == tb && j < tid)) ? 1 : 0;
            }
            tsrt[rk]  = tb;
            order[rk] = tid;
        }
        for (int i = tid; i < B_ * H_; i += NT) {
            int b = i >> 8, k = i & 255;
            a0[b][k] = (__bf16)input[(size_t)b * T_ * H_ + (size_t)tstep * H_ + k];
            a1[b][k] = (__bf16)hglob[i];
        }
        __syncthreads();
        if (tid < B_ - 1) dts[tid] = tsrt[tid + 1] - tsrt[tid];

        // ---- phase B: GRU cell (wave = N-strip, loop M blocks, 6 fused accums) ----
        const float bir = b_ih[col], biz = b_ih[col + 256], bin = b_ih[col + 512];
        const float bhr = b_hh[col], bhz = b_hh[col + 256], bhn = b_hh[col + 512];
#pragma clang loop unroll(disable)
        for (int m0 = 0; m0 < 4; ++m0) {
            v8f air{}, aiz{}, ain{}, ahr{}, ahz{}, ahn{};
#pragma clang loop unroll(disable)
            for (int kb = 0; kb < 8; ++kb) {
                v16bf ax = load_a(&a0[0][0], m0 * 16 + lcol, kb, khalf);
                v16bf ah = load_a(&a1[0][0], m0 * 16 + lcol, kb, khalf);
                v16bf br = load_b(Wih, col, kb, khalf);
                v16bf bz = load_b(Wih, 256 + col, kb, khalf);
                v16bf bn = load_b(Wih, 512 + col, kb, khalf);
                air = wmma_bf16(ax, br, air);
                aiz = wmma_bf16(ax, bz, aiz);
                ain = wmma_bf16(ax, bn, ain);
                br = load_b(Whh, col, kb, khalf);
                bz = load_b(Whh, 256 + col, kb, khalf);
                bn = load_b(Whh, 512 + col, kb, khalf);
                ahr = wmma_bf16(ah, br, ahr);
                ahz = wmma_bf16(ah, bz, ahz);
                ahn = wmma_bf16(ah, bn, ahn);
            }
#pragma unroll
            for (int r = 0; r < 8; ++r) {
                const int mrow = m0 * 16 + r + khalf * 8;   // C layout: lanes16-31 -> M+8
                float rr = sigmoidf_(air[r] + bir + ahr[r] + bhr);
                float zz = sigmoidf_(aiz[r] + biz + ahz[r] + bhz);
                float nn = tanhf_(ain[r] + bin + rr * (ahn[r] + bhn));
                float hold = hglob[mrow * H_ + col];
                float hnew = (1.0f - zz) * nn + zz * hold;
                hglob[mrow * H_ + col] = hnew;                 // traj[0] default
                y[mrow][col] = hnew;                           // ODE init
                out[(size_t)mrow * T_ * H_ + (size_t)tstep * H_ + col] = hnew;
            }
        }
        __syncthreads();

        // ---- phase C: 63 Euler steps; 3 barriers per step ----
        const float bb1 = b1[col], bb2 = b2[col], bb3 = b3[col];
        for (int s = 0; s < B_ - 1; ++s) {
            // capture snapshot s (y holds traj[s]); exact under ties since dt==0
            if (s > 0 && tid < H_) {
                int b0 = order[s];
                hglob[b0 * H_ + tid] = y[b0][tid];
            }
            // stage1: a1 = bf16(softplus(y @ W1^T + b1)); A f32->bf16 on the fly, B from LDS
            {
                v8f acc[4] = {v8f{}, v8f{}, v8f{}, v8f{}};
                gemm_strip<true>(&y[0][0], &W1lds[0][0], lcol, khalf, col, acc);
#pragma unroll
                for (int m = 0; m < 4; ++m)
#pragma unroll
                    for (int r = 0; r < 8; ++r)
                        a1[m * 16 + r + khalf * 8][col] = (__bf16)softplusf_(acc[m][r] + bb1);
            }
            __syncthreads();

            // stage2: a0 = bf16(softplus(h1 @ W2^T + b2)); B streamed from L2
            {
                v8f acc[4] = {v8f{}, v8f{}, v8f{}, v8f{}};
                gemm_strip<false>(&a1[0][0], W2, lcol, khalf, col, acc);
#pragma unroll
                for (int m = 0; m < 4; ++m)
#pragma unroll
                    for (int r = 0; r < 8; ++r)
                        a0[m * 16 + r + khalf * 8][col] = (__bf16)softplusf_(acc[m][r] + bb2);
            }
            __syncthreads();

            // stage3: y += (h2 @ W3^T + b3) * dt   (f32 state update in LDS)
            {
                const float dt = dts[s];
                v8f acc[4] = {v8f{}, v8f{}, v8f{}, v8f{}};
                gemm_strip<false>(&a0[0][0], W3, lcol, khalf, col, acc);
#pragma unroll
                for (int m = 0; m < 4; ++m)
#pragma unroll
                    for (int r = 0; r < 8; ++r)
                        y[m * 16 + r + khalf * 8][col] += (acc[m][r] + bb3) * dt;
            }
            __syncthreads();
        }
        // final snapshot: traj[63]
        if (tid < H_) {
            int b0 = order[B_ - 1];
            hglob[b0 * H_ + tid] = y[b0][tid];
        }
        __syncthreads();
    }
}

extern "C" void kernel_launch(void* const* d_in, const int* in_sizes, int n_in,
                              void* d_out, int out_size, void* d_ws, size_t ws_size,
                              hipStream_t stream) {
    const float* input = (const float*)d_in[0];
    const float* times = (const float*)d_in[1];
    const float* W_ih  = (const float*)d_in[2];
    const float* W_hh  = (const float*)d_in[3];
    const float* b_ih  = (const float*)d_in[4];
    const float* b_hh  = (const float*)d_in[5];
    const float* oW1   = (const float*)d_in[6];
    const float* ob1   = (const float*)d_in[7];
    const float* oW2   = (const float*)d_in[8];
    const float* ob2   = (const float*)d_in[9];
    const float* oW3   = (const float*)d_in[10];
    const float* ob3   = (const float*)d_in[11];

    // d_ws layout (bf16 elements): Wih[196608] Whh[196608] W1[65536] W2[65536] W3[65536],
    // then f32 h[64*256]. Total ~1.22 MB.
    __bf16* wbf   = (__bf16*)d_ws;
    float*  hglob = (float*)((char*)d_ws + (size_t)589824 * sizeof(__bf16));

    const int total_w = 589824;
    convert_weights_kernel<<<(total_w + 255) / 256, 256, 0, stream>>>(
        W_ih, W_hh, oW1, oW2, oW3, wbf);

    gru_ode_kernel<<<1, NT, 0, stream>>>(
        input, times, b_ih, b_hh, ob1, ob2, ob3,
        wbf,                 // Wih
        wbf + 196608,        // Whh
        wbf + 393216,        // W1
        wbf + 458752,        // W2
        wbf + 524288,        // W3
        hglob, (float*)d_out);
}
